// MultiHeadSelfAttention_76630806495864
// MI455X (gfx1250) — matrix operand order repaired and checked by reading, not compiled, not run on hardware
//
#include <hip/hip_runtime.h>
#include <hip/hip_bf16.h>

// ---- problem dims ----
#define EMBED 1024
#define HEADS 16
#define HDIM  64
#define SEQ   2048
#define BATCH 2
#define NTOK  (BATCH * SEQ)

#if defined(__gfx1250__) && __has_builtin(__builtin_amdgcn_tensor_load_to_lds)
#define HAVE_TDM 1
#else
#define HAVE_TDM 0
#endif

typedef __attribute__((ext_vector_type(16))) __bf16 v16bf;
typedef __attribute__((ext_vector_type(8)))  __bf16 v8bf;
typedef __attribute__((ext_vector_type(4)))  __bf16 v4bf;
typedef __attribute__((ext_vector_type(8)))  float  v8f;
typedef __attribute__((ext_vector_type(4)))  unsigned int u32x4;
typedef __attribute__((ext_vector_type(8)))  int          i32x8;
typedef __attribute__((ext_vector_type(4)))  int          i32x4;

static __device__ __forceinline__ v8f wmma_bf16(v16bf a, v16bf b, v8f c) {
  return __builtin_amdgcn_wmma_f32_16x16x32_bf16(
      /*neg_a=*/false, a, /*neg_b=*/false, b,
      /*c_mod=*/(short)0, c, /*reuse_a=*/false, /*reuse_b=*/false);
}

// ---- Tensor Data Mover: DMA one 2D tile (nrows x row_units, 8B units) ----
// D# per CDNA5 ISA: group0 = {count, lds_addr,
// global_addr, type=2}; group1 = {data_size=8B, tensor_dim0/1, tile_dim0/1,
// tensor_dim0_stride}; groups 2/3 zero (2D tensor).
static __device__ __forceinline__ void tdm_load_2d(const void* gptr, void* lds,
                                                   unsigned row_units,
                                                   unsigned nrows,
                                                   unsigned stride_units) {
#if HAVE_TDM
  unsigned long long ga = (unsigned long long)(uintptr_t)gptr;
  u32x4 g0;
  g0[0] = 1u;                                                 // count=1 (valid)
  g0[1] = (unsigned)(uintptr_t)lds;                           // lds byte addr
  g0[2] = (unsigned)(ga & 0xFFFFFFFFu);                       // global_addr lo
  g0[3] = (unsigned)((ga >> 32) & 0x01FFFFFFu) | (2u << 30);  // hi + type=2
  i32x8 g1;
  g1[0] = (int)(3u << 16);                                    // data_size=8B
  g1[1] = (int)((row_units & 0xFFFFu) << 16);                 // tensor_dim0 lo16
  g1[2] = (int)((row_units >> 16) | ((nrows & 0xFFFFu) << 16)); // d0 hi|d1 lo
  g1[3] = (int)((nrows >> 16) | ((row_units & 0xFFFFu) << 16)); // d1 hi|tile_dim0
  g1[4] = (int)(nrows & 0xFFFFu);                             // tile_dim1
  g1[5] = (int)stride_units;                                  // dim0_stride lo32
  g1[6] = 0;
  g1[7] = 0;
  i32x4 z = {0, 0, 0, 0};
#if __clang_major__ >= 23
  i32x8 z8 = {0, 0, 0, 0, 0, 0, 0, 0};
  __builtin_amdgcn_tensor_load_to_lds(g0, g1, z, z, z8, 0);
#else
  __builtin_amdgcn_tensor_load_to_lds(g0, g1, z, z, 0);
#endif
#else
  (void)gptr; (void)lds; (void)row_units; (void)nrows; (void)stride_units;
#endif
}

static __device__ __forceinline__ void wait_tensor0() {
#if HAVE_TDM
  __builtin_amdgcn_s_wait_tensorcnt(0);
#endif
}
static __device__ __forceinline__ void wait_tensor2() {
#if HAVE_TDM
  __builtin_amdgcn_s_wait_tensorcnt(2);
#endif
}

// ---------------- f32 -> bf16 cast (vectorized x4) ----------------
__global__ void cast_f32_bf16_kernel(const float* __restrict__ in,
                                     __bf16* __restrict__ out, int n4) {
  int i = blockIdx.x * blockDim.x + threadIdx.x;
  if (i >= n4) return;
  float4 f = ((const float4*)in)[i];
  v4bf o;
  o[0] = (__bf16)f.x; o[1] = (__bf16)f.y; o[2] = (__bf16)f.z; o[3] = (__bf16)f.w;
  *(v4bf*)(out + 4 * (size_t)i) = o;
}

static __device__ __forceinline__ v16bf make_a(const __bf16* row, int half) {
  v8bf alo = *(const v8bf*)&row[half * 16];
  v8bf ahi = *(const v8bf*)&row[half * 16 + 8];
  return __builtin_shufflevector(alo, ahi, 0, 1, 2, 3, 4, 5, 6, 7,
                                 8, 9, 10, 11, 12, 13, 14, 15);
}

// ---- WMMA macro-tile: 32(M-per-wave, 2 groups) x 64(N) x 32(K) ----
static __device__ __forceinline__ void compute_tile_128x64(
    const __bf16 (*__restrict__ As)[32], const __bf16 (*__restrict__ Bs)[32],
    v8f (*__restrict__ acc)[4], int wave, int half, int l16) {
  v16bf a[2];
#pragma unroll
  for (int g = 0; g < 2; ++g)
    a[g] = make_a(&As[wave * 32 + g * 16 + l16][0], half);
#pragma unroll
  for (int t = 0; t < 4; ++t) {
    v16bf b = *(const v16bf*)&Bs[t * 16 + l16][half * 16];
#pragma unroll
    for (int g = 0; g < 2; ++g) acc[g][t] = wmma_bf16(a[g], b, acc[g][t]);
  }
}

// ---------------- bf16 WMMA GEMM:  C[m,n] = A[m,:] . W[n,:] + bias[n] ----
// Block: 128 threads (4 waves). Block tile 128(M)x64(N), K-step 32,
// TDM double-buffered LDS.  Wave owns 32 rows -> 8 WMMA per K-step.
// mode 0: bf16 -> [B,H,S,D]; mode 1: bf16 -> [B,H,D,S]; mode 2: f32 row-major.
__global__ __launch_bounds__(128)
void gemm_bf16_kernel(const __bf16* __restrict__ A,   // [M, K] row-major
                      const __bf16* __restrict__ W,   // [N, K] row-major
                      const float*  __restrict__ bias,// [N]
                      __bf16* __restrict__ out_bf,
                      float*  __restrict__ out_f32,
                      int M, int K, int mode) {
#if HAVE_TDM
  __shared__ __align__(16) __bf16 As[2][128][32];
  __shared__ __align__(16) __bf16 Bs[2][64][32];
#else
  __shared__ __align__(16) __bf16 As[1][128][32];
  __shared__ __align__(16) __bf16 Bs[1][64][32];
#endif
  const int m0   = blockIdx.x * 128;
  const int n0   = blockIdx.y * 64;
  const int tid  = threadIdx.x;
  const int wave = tid >> 5;
  const int lane = tid & 31;
  const int half = lane >> 4;
  const int l16  = lane & 15;

  v8f acc[2][4] = {};

#if HAVE_TDM
  const unsigned strideU = (unsigned)(K >> 2);  // row stride in 8B units
  if (wave == 0) {
    tdm_load_2d(A + (size_t)m0 * K, &As[0][0][0], 8, 128, strideU);
    tdm_load_2d(W + (size_t)n0 * K, &Bs[0][0][0], 8, 64, strideU);
  }
  int buf = 0;
  for (int k0 = 0; k0 < K; k0 += 32, buf ^= 1) {
    if (wave == 0) {
      if (k0 + 32 < K) {  // prefetch next tiles into other buffer
        tdm_load_2d(A + (size_t)m0 * K + k0 + 32, &As[buf ^ 1][0][0], 8, 128, strideU);
        tdm_load_2d(W + (size_t)n0 * K + k0 + 32, &Bs[buf ^ 1][0][0], 8, 64, strideU);
        wait_tensor2();  // TDM completes in-order: current 2 tiles done
      } else {
        wait_tensor0();
      }
    }
    __syncthreads();
    compute_tile_128x64(As[buf], Bs[buf], acc, wave, half, l16);
    __syncthreads();  // reads done before next TDM overwrites this buffer
  }
#else
  for (int k0 = 0; k0 < K; k0 += 32) {
    for (int i = tid; i < 128 * 32 / 8; i += 128) {
      int r = i >> 2, c = (i & 3) << 3;
      *(uint4*)&As[0][r][c] = *(const uint4*)&A[(size_t)(m0 + r) * K + k0 + c];
    }
    for (int i = tid; i < 64 * 32 / 8; i += 128) {
      int r = i >> 2, c = (i & 3) << 3;
      *(uint4*)&Bs[0][r][c] = *(const uint4*)&W[(size_t)(n0 + r) * K + k0 + c];
    }
    __syncthreads();
    compute_tile_128x64(As[0], Bs[0], acc, wave, half, l16);
    __syncthreads();
  }
#endif

  // Epilogue. C layout: VGPR r -> row m = 32*wave + 16*g + 8*half + r,
  // col n = 16*t + l16.
#pragma unroll
  for (int t = 0; t < 4; ++t) {
    const int n = n0 + t * 16 + l16;
    const float bv = bias[n];
#pragma unroll
    for (int g = 0; g < 2; ++g) {
#pragma unroll
      for (int r = 0; r < 8; ++r) {
        const int m = m0 + wave * 32 + g * 16 + half * 8 + r;
        const float val = acc[g][t][r] + bv;
        if (mode == 2) {
          out_f32[(size_t)m * EMBED + n] = val;
        } else {
          const int b = m / SEQ, s = m % SEQ, h = n / HDIM, d = n % HDIM;
          size_t o;
          if (mode == 0) o = (((size_t)b * HEADS + h) * SEQ + s) * HDIM + d; // [B,H,S,D]
          else           o = (((size_t)b * HEADS + h) * HDIM + d) * SEQ + s; // [B,H,D,S]
          out_bf[o] = (__bf16)val;
        }
      }
    }
  }
}

// ---------------- flash attention (bf16 WMMA, f32 online softmax) --------
// Grid: (B*H, SEQ/64). Block: 128 threads (4 waves); wave w owns 16 q-rows.
__global__ __launch_bounds__(128)
void attn_kernel(const __bf16* __restrict__ Q,   // [B,H,S,D]
                 const __bf16* __restrict__ Km,  // [B,H,S,D]
                 const __bf16* __restrict__ Vt,  // [B,H,D,S]
                 __bf16* __restrict__ Ctx) {     // [B,S,H,D] == [NTOK,EMBED]
  __shared__ __align__(16) __bf16 Qs[64][HDIM];
#if HAVE_TDM
  __shared__ __align__(16) __bf16 Ks[2][64][HDIM];
  __shared__ __align__(16) __bf16 Vs[2][HDIM][64];
#else
  __shared__ __align__(16) __bf16 Ks[1][64][HDIM];
  __shared__ __align__(16) __bf16 Vs[1][HDIM][64];
#endif
  __shared__ __align__(16) __bf16 Ps[64][64];    // [q][k]

  const int bh   = blockIdx.x;
  const int q0   = blockIdx.y * 64;
  const int tid  = threadIdx.x;
  const int wave = tid >> 5;
  const int lane = tid & 31;
  const int half = lane >> 4;
  const int l16  = lane & 15;

  const __bf16* Qbh = Q  + (size_t)bh * SEQ * HDIM;
  const __bf16* Kbh = Km + (size_t)bh * SEQ * HDIM;
  const __bf16* Vbh = Vt + (size_t)bh * HDIM * SEQ;

#if HAVE_TDM
  if (wave == 0) {
    // Q tile: 64x64 bf16 contiguous (8 KB) -> 1D DMA of 1024 x 8B
    tdm_load_2d(Qbh + (size_t)q0 * HDIM, &Qs[0][0], 1024, 1, 1024);
    // K tile: also contiguous rows [k0..k0+63] x D
    tdm_load_2d(Kbh, &Ks[0][0][0], 1024, 1, 1024);
    // V^T tile: 64 rows (d) of 64 keys, row stride SEQ -> 16 x 64, stride 512
    tdm_load_2d(Vbh, &Vs[0][0][0], 16, 64, SEQ / 4);
  }
#else
  for (int i = tid; i < 64 * HDIM / 8; i += 128) {
    int r = i >> 3, c = (i & 7) * 8;
    *(uint4*)&Qs[r][c] = *(const uint4*)&Qbh[(size_t)(q0 + r) * HDIM + c];
  }
#endif

  v8f octx[4] = {};
  float row_max[8], row_sum[8];
#pragma unroll
  for (int r = 0; r < 8; ++r) { row_max[r] = -1e30f; row_sum[r] = 0.f; }
  const float scale = 0.125f;  // 1/sqrt(64)

  int buf = 0;
  for (int k0 = 0; k0 < SEQ; k0 += 64, buf ^= 1) {
#if HAVE_TDM
    if (wave == 0) {
      if (k0 + 64 < SEQ) {
        tdm_load_2d(Kbh + (size_t)(k0 + 64) * HDIM, &Ks[buf ^ 1][0][0], 1024, 1, 1024);
        tdm_load_2d(Vbh + (k0 + 64), &Vs[buf ^ 1][0][0], 16, 64, SEQ / 4);
        wait_tensor2();  // Q + current K/V done (in-order completion)
      } else {
        wait_tensor0();
      }
    }
    __syncthreads();
#else
    __syncthreads();
    for (int i = tid; i < 64 * HDIM / 8; i += 128) {
      int r = i >> 3, c = (i & 7) * 8;
      *(uint4*)&Ks[0][r][c] = *(const uint4*)&Kbh[(size_t)(k0 + r) * HDIM + c];
    }
    for (int i = tid; i < 64 * 64 / 8; i += 128) {
      int r = i >> 3, c = (i & 7) * 8;
      *(uint4*)&Vs[0][r][c] = *(const uint4*)&Vbh[(size_t)r * SEQ + k0 + c];
    }
    __syncthreads();
#endif

    // ---- S = Q K^T (reduction dim = d = 64) ----
    v8f sacc[4] = {};
#pragma unroll
    for (int kk = 0; kk < HDIM; kk += 32) {
      v16bf a = make_a(&Qs[wave * 16 + l16][kk], half);
#pragma unroll
      for (int t = 0; t < 4; ++t) {
        v16bf b = *(const v16bf*)&Ks[buf][t * 16 + l16][kk + half * 16];
        sacc[t] = wmma_bf16(a, b, sacc[t]);
      }
    }

    // ---- online softmax (a q-row lives in one 16-lane half) ----
#pragma unroll
    for (int r = 0; r < 8; ++r) {
      float mx = fmaxf(fmaxf(sacc[0][r], sacc[1][r]),
                       fmaxf(sacc[2][r], sacc[3][r])) * scale;
#pragma unroll
      for (int off = 8; off; off >>= 1)
        mx = fmaxf(mx, __shfl_xor(mx, off, 16));
      const float nm = fmaxf(row_max[r], mx);
      const float al = __expf(row_max[r] - nm);
      row_max[r] = nm;
      float ps = 0.f;
#pragma unroll
      for (int t = 0; t < 4; ++t) {
        float p = __expf(sacc[t][r] * scale - nm);
        sacc[t][r] = p;
        ps += p;
      }
#pragma unroll
      for (int off = 8; off; off >>= 1)
        ps += __shfl_xor(ps, off, 16);
      row_sum[r] = row_sum[r] * al + ps;
#pragma unroll
      for (int t = 0; t < 4; ++t) octx[t][r] *= al;
    }

    // ---- P (C-layout) -> LDS -> A-layout for P·V ----
#pragma unroll
    for (int t = 0; t < 4; ++t)
#pragma unroll
      for (int r = 0; r < 8; ++r)
        Ps[wave * 16 + half * 8 + r][t * 16 + l16] = (__bf16)sacc[t][r];
    __syncthreads();

    // ---- ctx += P · V (reduction dim = k = 64) ----
#pragma unroll
    for (int kk = 0; kk < 64; kk += 32) {
      v16bf a = make_a(&Ps[wave * 16 + l16][kk], half);
#pragma unroll
      for (int t = 0; t < 4; ++t) {
        v16bf b = *(const v16bf*)&Vs[buf][t * 16 + l16][kk + half * 16];
        octx[t] = wmma_bf16(a, b, octx[t]);
      }
    }
    __syncthreads();  // all reads of this buffer done before next TDM/copy
  }

  // ---- normalize + store ctx as [B,S,H,D] (== [NTOK, EMBED] row-major) ----
  const int b = bh / HEADS, h = bh % HEADS;
#pragma unroll
  for (int t = 0; t < 4; ++t) {
    const int d = t * 16 + l16;
#pragma unroll
    for (int r = 0; r < 8; ++r) {
      const int s = q0 + wave * 16 + half * 8 + r;
      Ctx[(((size_t)b * SEQ + s) * HEADS + h) * HDIM + d] =
          (__bf16)(octx[t][r] / row_sum[r]);
    }
  }
}

// ---------------- host launch ----------------
extern "C" void kernel_launch(void* const* d_in, const int* in_sizes, int n_in,
                              void* d_out, int out_size, void* d_ws, size_t ws_size,
                              hipStream_t stream) {
  const float* x  = (const float*)d_in[0];
  const float* wq = (const float*)d_in[1];
  const float* bq = (const float*)d_in[2];
  const float* wk = (const float*)d_in[3];
  const float* bk = (const float*)d_in[4];
  const float* wv = (const float*)d_in[5];
  const float* bv = (const float*)d_in[6];
  const float* wo = (const float*)d_in[7];
  const float* bo = (const float*)d_in[8];
  float* out = (float*)d_out;

  // workspace carve-out (bytes): 48 MiB total
  char* ws = (char*)d_ws;
  const size_t MB = 1u << 20;
  __bf16* xb   = (__bf16*)(ws + 0);        //  8 MiB  [NTOK,EMBED]
  __bf16* wqb  = (__bf16*)(ws + 8  * MB);  //  2 MiB
  __bf16* wkb  = (__bf16*)(ws + 10 * MB);  //  2 MiB
  __bf16* wvb  = (__bf16*)(ws + 12 * MB);  //  2 MiB
  __bf16* wob  = (__bf16*)(ws + 14 * MB);  //  2 MiB
  __bf16* qb   = (__bf16*)(ws + 16 * MB);  //  8 MiB  [B,H,S,D]
  __bf16* kb   = (__bf16*)(ws + 24 * MB);  //  8 MiB  [B,H,S,D]
  __bf16* vtb  = (__bf16*)(ws + 32 * MB);  //  8 MiB  [B,H,D,S]
  __bf16* ctxb = (__bf16*)(ws + 40 * MB);  //  8 MiB  [NTOK,EMBED]

  // casts
  cast_f32_bf16_kernel<<<(NTOK * EMBED / 4 + 255) / 256, 256, 0, stream>>>(x, xb, NTOK * EMBED / 4);
  cast_f32_bf16_kernel<<<(EMBED * EMBED / 4 + 255) / 256, 256, 0, stream>>>(wq, wqb, EMBED * EMBED / 4);
  cast_f32_bf16_kernel<<<(EMBED * EMBED / 4 + 255) / 256, 256, 0, stream>>>(wk, wkb, EMBED * EMBED / 4);
  cast_f32_bf16_kernel<<<(EMBED * EMBED / 4 + 255) / 256, 256, 0, stream>>>(wv, wvb, EMBED * EMBED / 4);
  cast_f32_bf16_kernel<<<(EMBED * EMBED / 4 + 255) / 256, 256, 0, stream>>>(wo, wob, EMBED * EMBED / 4);

  // projections (block tile 128x64)
  dim3 gg(NTOK / 128, EMBED / 64);
  gemm_bf16_kernel<<<gg, 128, 0, stream>>>(xb, wqb, bq, qb,  nullptr, NTOK, EMBED, 0);
  gemm_bf16_kernel<<<gg, 128, 0, stream>>>(xb, wkb, bk, kb,  nullptr, NTOK, EMBED, 0);
  gemm_bf16_kernel<<<gg, 128, 0, stream>>>(xb, wvb, bv, vtb, nullptr, NTOK, EMBED, 1);

  // attention
  attn_kernel<<<dim3(BATCH * HEADS, SEQ / 64), 128, 0, stream>>>(qb, kb, vtb, ctxb);

  // output projection -> f32 d_out
  gemm_bf16_kernel<<<gg, 128, 0, stream>>>(ctxb, wob, bo, nullptr, out, NTOK, EMBED, 2);
}
